// EntityPairAttentionNeighboursRelationEmbedding_45397804318893
// MI455X (gfx1250) — matrix operand order repaired and checked by reading, not compiled, not run on hardware
//
#include <hip/hip_runtime.h>

// Problem constants (match reference)
#define NPAIRS 4096   // N_OUTER * M_INNER
#define LMAX   256    // padded neighbour list length
#define KDIM   256    // relation embedding dim

typedef __attribute__((ext_vector_type(16))) _Float16 v16h;
typedef __attribute__((ext_vector_type(8)))  _Float16 v8h;
typedef __attribute__((ext_vector_type(8)))  float    v8f;

// ---------------------------------------------------------------------------
// Pre-pass: convert the relation table f32 -> f16 into workspace.
// Streams ~77 MB through HBM once (~3us at 23.3 TB/s); the main kernel then
// gathers half the bytes per row with zero converts in its hot loop.
// ---------------------------------------------------------------------------
__global__ __launch_bounds__(256)
void cvt_table_f16_kernel(const float* __restrict__ src,
                          _Float16* __restrict__ dst, int n8)
{
    const int i = blockIdx.x * blockDim.x + threadIdx.x;
    if (i >= n8) return;
    const float4* p = reinterpret_cast<const float4*>(src) + 2 * (size_t)i;
    const float4 a = p[0];
    const float4 b = p[1];
    v8h h;
    h[0] = (_Float16)a.x; h[1] = (_Float16)a.y;
    h[2] = (_Float16)a.z; h[3] = (_Float16)a.w;
    h[4] = (_Float16)b.x; h[5] = (_Float16)b.y;
    h[6] = (_Float16)b.z; h[7] = (_Float16)b.w;
    *(reinterpret_cast<v8h*>(dst) + i) = h;
}

// ---------------------------------------------------------------------------
// One workgroup (256 threads = 8 wave32) per entity pair n.
// Phase 1: masked softmax over w1+w2 (block reduction in LDS); neighbour/cand
//          index gathers are issued BEFORE phase 1 so their latency hides
//          under the reduction tree.
// Phase 2: out[n] = sum_l w[l] * (table[neigh[l]] . table[cand]) via
//          v_wmma_f32_16x16x32_f16. Each wave owns 2 l-tiles processed
//          concurrently with independent accumulator chains (no WMMA->WMMA
//          RAW between consecutive wmmas), sharing hoisted B fragments.
// ---------------------------------------------------------------------------
template <bool USE_H>
__global__ __launch_bounds__(256)
void ep_attn_wmma_kernel(const float*    __restrict__ table,
                         const _Float16* __restrict__ tableH,
                         const float*    __restrict__ w1,
                         const float*    __restrict__ w2,
                         const int*      __restrict__ cand_idx,
                         const int*      __restrict__ neigh_idx,
                         const int*      __restrict__ lengths,
                         float*          __restrict__ out)
{
    __shared__ float sRed[LMAX];
    __shared__ float sW[LMAX];
    __shared__ __attribute__((aligned(32))) _Float16 sCandH[KDIM];
    __shared__ float sOut;

    const int n   = blockIdx.x;
    const int tid = threadIdx.x;

    const int lane   = tid & 31;   // wave32 lane
    const int wave   = tid >> 5;   // 8 waves
    const int laneM  = lane & 15;  // A-matrix row / D column index
    const int laneHi = lane >> 4;  // K-half selector

    // Issue all index gathers up front; latency overlaps the softmax phase.
    const int lbase0 = wave * 16;            // first l-tile of this wave
    const int lbase1 = (wave + 8) * 16;      // second l-tile of this wave
    const int ridx0  = neigh_idx[(size_t)n * LMAX + lbase0 + laneM];
    const int ridx1  = neigh_idx[(size_t)n * LMAX + lbase1 + laneM];
    const size_t crow = (size_t)cand_idx[n] * KDIM;
    const int len     = lengths[n];

    // ---------------- Phase 1: masked softmax over L ----------------
    const float lw = (tid < len)
        ? (w1[(size_t)n * LMAX + tid] + w2[(size_t)n * LMAX + tid])
        : -1e30f;

    sRed[tid] = lw;
    __syncthreads();
    #pragma unroll
    for (int s = 128; s > 0; s >>= 1) {
        if (tid < s) sRed[tid] = fmaxf(sRed[tid], sRed[tid + s]);
        __syncthreads();
    }
    const float mx = sRed[0];
    __syncthreads();

    const float ev = (tid < len) ? __expf(lw - mx) : 0.0f;
    sRed[tid] = ev;
    __syncthreads();
    #pragma unroll
    for (int s = 128; s > 0; s >>= 1) {
        if (tid < s) sRed[tid] += sRed[tid + s];
        __syncthreads();
    }
    const float inv = 1.0f / sRed[0];   // len >= 1 so sum > 0
    sW[tid] = ev * inv;

    // Stage candidate embedding as f16 in LDS (one element per thread).
    if (USE_H) sCandH[tid] = tableH[crow + tid];
    else       sCandH[tid] = (_Float16)table[crow + tid];
    if (tid == 0) sOut = 0.0f;
    __syncthreads();

    // ---------------- Phase 2: dual-tile WMMA dot products ----------------
    // EXEC is all-ones through every WMMA (ISA requirement): no divergence
    // until the final per-column atomic.
    v8f acc0 = {};
    v8f acc1 = {};

    if (USE_H) {
        const _Float16* __restrict__ row0 = tableH + (size_t)ridx0 * KDIM;
        const _Float16* __restrict__ row1 = tableH + (size_t)ridx1 * KDIM;
        #pragma unroll
        for (int c = 0; c < 8; ++c) {          // K = 256 in chunks of 32
            const int k0 = c * 32 + laneHi * 8;
            // ISA 16-bit A layout: elems 0..7 -> K=k0..k0+7,
            //                      elems 8..15 -> K=k0+16..k0+23
            const v8h lo0 = *reinterpret_cast<const v8h*>(row0 + k0);
            const v8h hi0 = *reinterpret_cast<const v8h*>(row0 + k0 + 16);
            const v8h lo1 = *reinterpret_cast<const v8h*>(row1 + k0);
            const v8h hi1 = *reinterpret_cast<const v8h*>(row1 + k0 + 16);
            v16h a0, a1;
            #pragma unroll
            for (int e = 0; e < 8; ++e) {
                a0[e] = lo0[e]; a0[8 + e] = hi0[e];
                a1[e] = lo1[e]; a1[8 + e] = hi1[e];
            }
            const v8h blo = *reinterpret_cast<const v8h*>(&sCandH[c * 32 + laneHi * 8]);
            const v8h bhi = *reinterpret_cast<const v8h*>(&sCandH[c * 32 + laneHi * 8 + 16]);
            const v16h b = __builtin_shufflevector(blo, bhi, 0,1,2,3,4,5,6,7,8,9,10,11,12,13,14,15);
            acc0 = __builtin_amdgcn_wmma_f32_16x16x32_f16(
                false, a0, false, b, (short)0, acc0, false, false);
            acc1 = __builtin_amdgcn_wmma_f32_16x16x32_f16(
                false, a1, false, b, (short)0, acc1, false, false);
        }
    } else {
        const float* __restrict__ row0 = table + (size_t)ridx0 * KDIM;
        const float* __restrict__ row1 = table + (size_t)ridx1 * KDIM;
        #pragma unroll
        for (int c = 0; c < 8; ++c) {
            const int k0 = c * 32 + laneHi * 8;
            const float4 f0 = *reinterpret_cast<const float4*>(row0 + k0);
            const float4 f1 = *reinterpret_cast<const float4*>(row0 + k0 + 4);
            const float4 f2 = *reinterpret_cast<const float4*>(row0 + k0 + 16);
            const float4 f3 = *reinterpret_cast<const float4*>(row0 + k0 + 20);
            const float4 g0 = *reinterpret_cast<const float4*>(row1 + k0);
            const float4 g1 = *reinterpret_cast<const float4*>(row1 + k0 + 4);
            const float4 g2 = *reinterpret_cast<const float4*>(row1 + k0 + 16);
            const float4 g3 = *reinterpret_cast<const float4*>(row1 + k0 + 20);
            v16h a0, a1;
            a0[0]  = (_Float16)f0.x; a0[1]  = (_Float16)f0.y;
            a0[2]  = (_Float16)f0.z; a0[3]  = (_Float16)f0.w;
            a0[4]  = (_Float16)f1.x; a0[5]  = (_Float16)f1.y;
            a0[6]  = (_Float16)f1.z; a0[7]  = (_Float16)f1.w;
            a0[8]  = (_Float16)f2.x; a0[9]  = (_Float16)f2.y;
            a0[10] = (_Float16)f2.z; a0[11] = (_Float16)f2.w;
            a0[12] = (_Float16)f3.x; a0[13] = (_Float16)f3.y;
            a0[14] = (_Float16)f3.z; a0[15] = (_Float16)f3.w;
            a1[0]  = (_Float16)g0.x; a1[1]  = (_Float16)g0.y;
            a1[2]  = (_Float16)g0.z; a1[3]  = (_Float16)g0.w;
            a1[4]  = (_Float16)g1.x; a1[5]  = (_Float16)g1.y;
            a1[6]  = (_Float16)g1.z; a1[7]  = (_Float16)g1.w;
            a1[8]  = (_Float16)g2.x; a1[9]  = (_Float16)g2.y;
            a1[10] = (_Float16)g2.z; a1[11] = (_Float16)g2.w;
            a1[12] = (_Float16)g3.x; a1[13] = (_Float16)g3.y;
            a1[14] = (_Float16)g3.z; a1[15] = (_Float16)g3.w;

            const v8h blo = *reinterpret_cast<const v8h*>(&sCandH[c * 32 + laneHi * 8]);
            const v8h bhi = *reinterpret_cast<const v8h*>(&sCandH[c * 32 + laneHi * 8 + 16]);
            const v16h b = __builtin_shufflevector(blo, bhi, 0,1,2,3,4,5,6,7,8,9,10,11,12,13,14,15);
            acc0 = __builtin_amdgcn_wmma_f32_16x16x32_f16(
                false, a0, false, b, (short)0, acc0, false, false);
            acc1 = __builtin_amdgcn_wmma_f32_16x16x32_f16(
                false, a1, false, b, (short)0, acc1, false, false);
        }
    }

    // D layout: VGPR v holds row M = v + 8*laneHi, all 16 columns equal d[l].
    float s = 0.0f;
    #pragma unroll
    for (int v = 0; v < 8; ++v) {
        s += sW[lbase0 + laneHi * 8 + v] * acc0[v];
        s += sW[lbase1 + laneHi * 8 + v] * acc1[v];
    }
    if (laneM == 0)                      // column N==0 only: count each l once
        atomicAdd(&sOut, s);             // ds_add_f32

    __syncthreads();
    if (tid == 0) out[n] = sOut;
}

extern "C" void kernel_launch(void* const* d_in, const int* in_sizes, int n_in,
                              void* d_out, int out_size, void* d_ws, size_t ws_size,
                              hipStream_t stream) {
    (void)n_in; (void)out_size;
    const float* table     = (const float*)d_in[0];
    const float* w1        = (const float*)d_in[1];
    const float* w2        = (const float*)d_in[2];
    const int*   cand_idx  = (const int*)d_in[3];
    const int*   neigh_idx = (const int*)d_in[4];
    const int*   lengths   = (const int*)d_in[5];
    float*       out       = (float*)d_out;

    const size_t tab_elems = (size_t)in_sizes[0];        // R * K
    const size_t need      = tab_elems * sizeof(_Float16);

    if (d_ws != nullptr && ws_size >= need) {
        // Fast path: one-shot f32->f16 table conversion, then f16 gathers.
        _Float16* tableH = (_Float16*)d_ws;
        const int n8     = (int)(tab_elems / 8);
        const int blocks = (n8 + 255) / 256;
        cvt_table_f16_kernel<<<blocks, 256, 0, stream>>>(table, tableH, n8);
        ep_attn_wmma_kernel<true><<<NPAIRS, 256, 0, stream>>>(
            table, tableH, w1, w2, cand_idx, neigh_idx, lengths, out);
    } else {
        // Fallback: gather f32 rows and convert in-register.
        ep_attn_wmma_kernel<false><<<NPAIRS, 256, 0, stream>>>(
            table, nullptr, w1, w2, cand_idx, neigh_idx, lengths, out);
    }
}
